// BrainMoEMulti_86835648790905
// MI455X (gfx1250) — matrix operand order repaired and checked by reading, not compiled, not run on hardware
//
#include <hip/hip_runtime.h>
#include <hip/hip_bf16.h>

// ---------------- dims (match reference) ----------------
constexpr int B_   = 32;
constexpr int V_   = 15724;
constexpr int H_   = 1024;
constexpr int S_   = 77;
constexpr int D_   = 768;
constexpr int E0_  = 4;
constexpr int NB_  = 4;
constexpr int EF_  = 2;
constexpr int H0_  = 256;
constexpr int C0_  = 256;
constexpr int E1_  = 8;
constexpr int H1_  = 128;
constexpr int C1_  = 128;
constexpr int PH_  = 1536;
constexpr int OUT_ = S_ * D_;          // 59136
constexpr int KP_  = 15744;            // V_ padded to multiple of 32

// ---------------- types ----------------
typedef __attribute__((ext_vector_type(16))) __bf16        bf16x16;
typedef __attribute__((ext_vector_type(8)))  float         floatx8;
typedef __attribute__((ext_vector_type(4)))  unsigned int  u32x4;
typedef __attribute__((ext_vector_type(8)))  int           i32x8;
typedef __attribute__((ext_vector_type(4)))  int           i32x4;

union FragU { bf16x16 bf; u32x4 u[2]; };

// ---------------- helpers ----------------
__device__ inline unsigned short f2bf(float x) {
  unsigned int u = __float_as_uint(x);
  unsigned int r = u + 0x7FFFu + ((u >> 16) & 1u);   // RNE
  return (unsigned short)(r >> 16);
}

__device__ inline float geluf(float x) {             // exact GELU
  return 0.5f * x * (1.0f + erff(x * 0.70710678118654752f));
}

__device__ inline float blockReduceSum(float v) {
  __shared__ float s[8];
  __shared__ float res;
  int lane = threadIdx.x & 31, w = threadIdx.x >> 5;
  #pragma unroll
  for (int o = 16; o > 0; o >>= 1) v += __shfl_down(v, o, 32);
  __syncthreads();
  if (lane == 0) s[w] = v;
  __syncthreads();
  if (threadIdx.x == 0) {
    float t = 0.f;
    int nw = blockDim.x >> 5;
    for (int i = 0; i < nw; i++) t += s[i];
    res = t;
  }
  __syncthreads();
  return res;
}

// ---------------- Tensor Data Mover helpers ----------------
typedef const char __attribute__((address_space(3)))* lds_cptr_t;

__device__ inline unsigned lds_off(const void* p) {
  return (unsigned)(unsigned long long)(lds_cptr_t)p;
}

// 2D tile load Global->LDS via TDM.  Elements are 2 bytes (bf16).
//   ldsAddr      : byte offset in LDS where tile starts
//   gptr         : global address of tile start (row 0, col 0)
//   tileK        : tile dim0 (contiguous elements per row)
//   tileRows     : tile dim1 (rows)
//   rowLen       : tensor dim0 == dim0 stride (elements)
//   rowsRemain   : tensor dim1 (rows available; OOB rows are zero-filled)
__device__ inline void tdm_load_tile(unsigned ldsAddr, const unsigned short* gptr,
                                     unsigned tileK, unsigned tileRows,
                                     unsigned rowLen, unsigned rowsRemain)
{
  unsigned long long ga = (unsigned long long)(uintptr_t)gptr;
  u32x4 g0;
  g0[0] = 1u;                                              // count=1 (valid user D#)
  g0[1] = ldsAddr;                                         // lds_addr
  g0[2] = (unsigned)ga;                                    // global_addr[31:0]
  g0[3] = (unsigned)((ga >> 32) & 0x01FFFFFFu) | 0x80000000u; // addr[56:32] | type=2
  i32x8 g1;
  g1[0] = (int)(1u << 16);                                 // data_size=1 (2 bytes)
  g1[1] = (int)((rowLen & 0xFFFFu) << 16);                 // tensor_dim0[15:0]
  g1[2] = (int)(((rowLen >> 16) & 0xFFFFu) | ((rowsRemain & 0xFFFFu) << 16));
  g1[3] = (int)(((rowsRemain >> 16) & 0xFFFFu) | ((tileK & 0xFFFFu) << 16));
  g1[4] = (int)(tileRows & 0xFFFFu);                       // tile_dim1 (tile_dim2=0)
  g1[5] = (int)rowLen;                                     // tensor_dim0_stride[31:0]
  g1[6] = 0;                                               // stride[47:32], dim1_stride lo
  g1[7] = 0;
  i32x4 z4 = {0, 0, 0, 0};
#if defined(__clang_major__) && (__clang_major__ >= 23)
  i32x8 z8 = {0, 0, 0, 0, 0, 0, 0, 0};
  __builtin_amdgcn_tensor_load_to_lds(g0, g1, z4, z4, z8, 0);
#else
  __builtin_amdgcn_tensor_load_to_lds(g0, g1, z4, z4, 0);
#endif
}

// ======================================================================
// Direct-global bf16 WMMA GEMM (mode 1: residual LN-block epilogue).
// Block = 8 waves, tile 32x256, each wave 16x64 (4 accumulators).
// ======================================================================
__global__ __launch_bounds__(256) void wmma_gemm(
    const unsigned short* __restrict__ A,  size_t sA,
    const unsigned short* __restrict__ W,  size_t sW,
    const float* __restrict__ bias,        size_t sBias,
    const float* __restrict__ R,           size_t sR,
    float* __restrict__ C,                 size_t sC,
    int M, int N, int K, int mode)
{
  int lane = threadIdx.x & 31;
  int wave = threadIdx.x >> 5;
  int row0 = blockIdx.y * 32 + (wave & 1) * 16;
  int col0 = blockIdx.x * 256 + (wave >> 1) * 64;
  if (row0 >= M || col0 >= N) return;
  int e = blockIdx.z;

  const unsigned short* Ae = A + (size_t)e * sA;
  const unsigned short* We = W + (size_t)e * sW;
  const float* be = bias ? (bias + (size_t)e * sBias) : nullptr;

  int r  = lane & 15;
  int hb = lane >> 4;
  const unsigned short* arow = Ae + (size_t)(row0 + r) * K + hb * 16;
  const unsigned short* brow[4];
  #pragma unroll
  for (int t = 0; t < 4; t++)
    brow[t] = We + (size_t)(col0 + t * 16 + r) * K + hb * 16;

  floatx8 zf = {0.f,0.f,0.f,0.f,0.f,0.f,0.f,0.f};
  floatx8 acc[4] = {zf, zf, zf, zf};

  for (int k0 = 0; k0 < K; k0 += 32) {
    if ((k0 & 255) == 0) {
      __builtin_prefetch(arow + k0 + 512, 0, 1);
      __builtin_prefetch(brow[0] + k0 + 512, 0, 1);
    }
    // issue ALL fragment loads first so waits can be partial
    FragU a, b0, b1, b2, b3;
    a.u[0]  = *(const u32x4*)(arow + k0);
    a.u[1]  = *(const u32x4*)(arow + k0 + 8);
    b0.u[0] = *(const u32x4*)(brow[0] + k0);
    b0.u[1] = *(const u32x4*)(brow[0] + k0 + 8);
    b1.u[0] = *(const u32x4*)(brow[1] + k0);
    b1.u[1] = *(const u32x4*)(brow[1] + k0 + 8);
    b2.u[0] = *(const u32x4*)(brow[2] + k0);
    b2.u[1] = *(const u32x4*)(brow[2] + k0 + 8);
    b3.u[0] = *(const u32x4*)(brow[3] + k0);
    b3.u[1] = *(const u32x4*)(brow[3] + k0 + 8);
    acc[0] = __builtin_amdgcn_wmma_f32_16x16x32_bf16(false, a.bf, false, b0.bf, (short)0, acc[0], false, false);
    acc[1] = __builtin_amdgcn_wmma_f32_16x16x32_bf16(false, a.bf, false, b1.bf, (short)0, acc[1], false, false);
    acc[2] = __builtin_amdgcn_wmma_f32_16x16x32_bf16(false, a.bf, false, b2.bf, (short)0, acc[2], false, false);
    acc[3] = __builtin_amdgcn_wmma_f32_16x16x32_bf16(false, a.bf, false, b3.bf, (short)0, acc[3], false, false);
  }

  int cn = lane & 15;
  #pragma unroll
  for (int t = 0; t < 4; t++) {
    int cc = col0 + t * 16 + cn;
    #pragma unroll
    for (int j = 0; j < 8; j++) {
      int rr = row0 + hb * 8 + j;
      float v = acc[t][j];
      if (be) v += be[cc];
      if (mode == 2) {
        atomicAdd(&C[(size_t)rr * N + cc], v);
      } else {
        if (mode == 1) v = R[(size_t)e * sR + (size_t)rr * N + cc] + geluf(v);
        C[(size_t)e * sC + (size_t)rr * N + cc] = v;
      }
    }
  }
}

// ======================================================================
// TDM-staged bf16 WMMA GEMM (modes 0 / 2). Double-buffered LDS tiles:
//   A tile 32x32 bf16, B tile 256x32 bf16 (weights, [N,K] layout),
// loaded by wave 0 via tensor_load_to_lds, published with TENSORcnt wait
// + workgroup barrier, consumed via ds_load_b128 fragment reads.
// ======================================================================
__global__ __launch_bounds__(256) void wmma_gemm_tdm(
    const unsigned short* __restrict__ A,  size_t sA,
    const unsigned short* __restrict__ W,  size_t sW,
    const float* __restrict__ bias,        size_t sBias,
    float* __restrict__ C,                 size_t sC,
    int M, int N, int K, int mode)
{
  __shared__ unsigned short bufA[2][32][32];
  __shared__ unsigned short bufB[2][256][32];

  int lane = threadIdx.x & 31;
  int wave = threadIdx.x >> 5;
  int row0 = blockIdx.y * 32;
  int colB = blockIdx.x * 256;
  int e = blockIdx.z;

  const unsigned short* Ae = A + (size_t)e * sA + (size_t)row0 * K;
  const unsigned short* We = W + (size_t)e * sW + (size_t)colB * K;
  const float* be = bias ? (bias + (size_t)e * sBias) : nullptr;
  unsigned rowsRemain = (unsigned)(N - colB);   // TDM zero-fills OOB rows

  unsigned ldsA[2] = { lds_off(&bufA[0][0][0]), lds_off(&bufA[1][0][0]) };
  unsigned ldsB[2] = { lds_off(&bufB[0][0][0]), lds_off(&bufB[1][0][0]) };

  int r  = lane & 15;
  int hb = lane >> 4;
  int nsteps = K >> 5;

  if (wave == 0) {
    tdm_load_tile(ldsA[0], Ae, 32, 32, (unsigned)K, 32u);
    tdm_load_tile(ldsB[0], We, 32, 256, (unsigned)K, rowsRemain);
  }

  floatx8 zf = {0.f,0.f,0.f,0.f,0.f,0.f,0.f,0.f};
  floatx8 acc[4] = {zf, zf, zf, zf};

  for (int s = 0; s < nsteps; ++s) {
    int p = s & 1;
    if (wave == 0) __builtin_amdgcn_s_wait_tensorcnt(0);  // tile p landed
    __syncthreads();                                      // publish tile p
    if (wave == 0 && (s + 1) < nsteps) {                  // prefetch tile p^1
      int k0 = (s + 1) << 5;
      tdm_load_tile(ldsA[p ^ 1], Ae + k0, 32, 32, (unsigned)K, 32u);
      tdm_load_tile(ldsB[p ^ 1], We + k0, 32, 256, (unsigned)K, rowsRemain);
    }
    // fragments from LDS tile p
    FragU a, b0, b1, b2, b3;
    const unsigned short* ar = &bufA[p][(wave & 1) * 16 + r][hb * 16];
    a.u[0] = *(const u32x4*)(ar);
    a.u[1] = *(const u32x4*)(ar + 8);
    int bn = (wave >> 1) * 64 + r;
    b0.u[0] = *(const u32x4*)(&bufB[p][bn +  0][hb * 16]);
    b0.u[1] = *(const u32x4*)(&bufB[p][bn +  0][hb * 16 + 8]);
    b1.u[0] = *(const u32x4*)(&bufB[p][bn + 16][hb * 16]);
    b1.u[1] = *(const u32x4*)(&bufB[p][bn + 16][hb * 16 + 8]);
    b2.u[0] = *(const u32x4*)(&bufB[p][bn + 32][hb * 16]);
    b2.u[1] = *(const u32x4*)(&bufB[p][bn + 32][hb * 16 + 8]);
    b3.u[0] = *(const u32x4*)(&bufB[p][bn + 48][hb * 16]);
    b3.u[1] = *(const u32x4*)(&bufB[p][bn + 48][hb * 16 + 8]);
    acc[0] = __builtin_amdgcn_wmma_f32_16x16x32_bf16(false, a.bf, false, b0.bf, (short)0, acc[0], false, false);
    acc[1] = __builtin_amdgcn_wmma_f32_16x16x32_bf16(false, a.bf, false, b1.bf, (short)0, acc[1], false, false);
    acc[2] = __builtin_amdgcn_wmma_f32_16x16x32_bf16(false, a.bf, false, b2.bf, (short)0, acc[2], false, false);
    acc[3] = __builtin_amdgcn_wmma_f32_16x16x32_bf16(false, a.bf, false, b3.bf, (short)0, acc[3], false, false);
    __syncthreads();     // all reads of tile p done before it is re-filled
  }

  int cn = lane & 15;
  int col0 = colB + (wave >> 1) * 64;
  int rbase = row0 + (wave & 1) * 16 + hb * 8;
  #pragma unroll
  for (int t = 0; t < 4; t++) {
    int cc = col0 + t * 16 + cn;
    if (cc >= N) continue;
    #pragma unroll
    for (int j = 0; j < 8; j++) {
      int rr = rbase + j;
      float v = acc[t][j];
      if (be) v += be[cc];
      if (mode == 2) atomicAdd(&C[(size_t)rr * N + cc], v);
      else           C[(size_t)e * sC + (size_t)rr * N + cc] = v;
    }
  }
}

// ======================================================================
// Tiled transpose + f32 -> bf16: Wf[z][K,N] -> Wt[z][N,Kp] (zero K-pad)
// ======================================================================
__global__ __launch_bounds__(256) void transpose_cvt(
    const float* __restrict__ Wf, size_t sIn,
    unsigned short* __restrict__ Wt, size_t sOut,
    int K, int N, int Kp)
{
  __shared__ float tile[32][33];
  int e  = blockIdx.z;
  int k0 = blockIdx.y * 32, n0 = blockIdx.x * 32;
  int tx = threadIdx.x & 31, ty = threadIdx.x >> 5;
  const float* src = Wf + (size_t)e * sIn;
  unsigned short* dst = Wt + (size_t)e * sOut;
  for (int i = ty; i < 32; i += 8) {
    int k = k0 + i, n = n0 + tx;
    tile[i][tx] = (k < K && n < N) ? src[(size_t)k * N + n] : 0.f;
  }
  __syncthreads();
  for (int i = ty; i < 32; i += 8) {
    int n = n0 + i, k = k0 + tx;
    if (n < N && k < Kp) dst[(size_t)n * Kp + k] = f2bf(tile[tx][i]);
  }
}

__global__ void cvt_pad_kernel(const float* __restrict__ X,
                               unsigned short* __restrict__ Y,
                               int K, int Kp, long long n)
{
  long long i = (long long)blockIdx.x * blockDim.x + threadIdx.x;
  if (i >= n) return;
  long long r = i / Kp; int k = (int)(i % Kp);
  Y[i] = (k < K) ? f2bf(X[r * (long long)K + k]) : (unsigned short)0;
}

__global__ void cvt_scale_kernel(const float* __restrict__ X,
                                 const float* __restrict__ scale,
                                 unsigned short* __restrict__ Y,
                                 float* __restrict__ Yf,
                                 long long n, int rowlen)
{
  long long i = (long long)blockIdx.x * blockDim.x + threadIdx.x;
  if (i >= n) return;
  float v = X[i];
  if (scale) v *= scale[i / rowlen];
  Y[i] = f2bf(v);
  if (Yf) Yf[i] = v;
}

// ======================================================================
// LayerNorm(+optional exact GELU) -> bf16. One row per block.
// ======================================================================
__global__ __launch_bounds__(256) void ln_gelu_kernel(
    const float* __restrict__ X,
    const float* __restrict__ g, const float* __restrict__ b, size_t gStride,
    unsigned short* __restrict__ Y, int Dd, int rowsPerG, int applyGelu)
{
  int row = blockIdx.x;
  int grp = row / rowsPerG;
  const float* x = X + (size_t)row * Dd;
  float s1 = 0.f, s2 = 0.f;
  for (int d = threadIdx.x; d < Dd; d += blockDim.x) {
    float v = x[d]; s1 += v; s2 += v * v;
  }
  s1 = blockReduceSum(s1);
  s2 = blockReduceSum(s2);
  float mean = s1 / Dd;
  float var  = s2 / Dd - mean * mean;
  float rstd = rsqrtf(var + 1e-5f);
  const float* gg = g + (size_t)grp * gStride;
  const float* bb = b + (size_t)grp * gStride;
  unsigned short* y = Y + (size_t)row * Dd;
  for (int d = threadIdx.x; d < Dd; d += blockDim.x) {
    float v = (x[d] - mean) * rstd * gg[d] + bb[d];
    if (applyGelu) v = geluf(v);
    y[d] = f2bf(v);
  }
}

// ======================================================================
// Capacity-based top-1 router (matches reference semantics).
// ======================================================================
__global__ __launch_bounds__(256) void router_kernel(
    const float* __restrict__ x,
    const float* __restrict__ gw, const float* __restrict__ gb,
    float* __restrict__ sel, float* __restrict__ wout, float* lbacc,
    int Bsz, int T, int E, int Cap)
{
  extern __shared__ float sm[];
  float* probs = sm;
  int*   amax  = (int*)(sm + (size_t)T * E);
  int g = blockIdx.y, b = blockIdx.x;
  const float* xg = x + ((size_t)g * Bsz + b) * T;

  float psum[4] = {0.f,0.f,0.f,0.f};
  float fcnt[4] = {0.f,0.f,0.f,0.f};
  for (int t = threadIdx.x; t < T; t += blockDim.x) {
    float xv = xg[t];
    float l[4]; float mx = -1e30f;
    for (int e = 0; e < E; e++) { l[e] = xv * gw[g*E+e] + gb[g*E+e]; mx = fmaxf(mx, l[e]); }
    float se = 0.f;
    for (int e = 0; e < E; e++) { l[e] = __expf(l[e] - mx); se += l[e]; }
    float inv = 1.f / se;
    int am = 0; float best = -1.f;
    for (int e = 0; e < E; e++) {
      float p = l[e] * inv;
      probs[t*E+e] = p; psum[e] += p;
      if (p > best) { best = p; am = e; }
    }
    amax[t] = am; fcnt[am] += 1.f;
  }
  __syncthreads();
  for (int e = 0; e < E; e++) {
    float ps = blockReduceSum(psum[e]);
    float fc = blockReduceSum(fcnt[e]);
    if (threadIdx.x == 0) {
      wout[((size_t)g * E + e) * Bsz + b] = ps / T;
      if (lbacc) { atomicAdd(&lbacc[e], fc); atomicAdd(&lbacc[E + e], ps); }
    }
  }
  __syncthreads();
  if (threadIdx.x == 0) {
    int cnt[4] = {0,0,0,0};
    for (int t = 0; t < T; t++) {
      int e = amax[t]; cnt[e]++;
      if (cnt[e] <= Cap)
        sel[(((size_t)g * E + e) * Bsz + b) * Cap + (cnt[e] - 1)] = xg[t];
    }
  }
}

// ---------------- epilogues ----------------
__global__ void combine_backbone_kernel(const float* __restrict__ bb0,
                                        const float* __restrict__ bb1,
                                        float* __restrict__ out, long long n)
{
  long long i = (long long)blockIdx.x * blockDim.x + threadIdx.x;
  if (i >= n) return;
  long long b = i / OUT_, j = i % OUT_;
  float s = 0.f;
  for (int e = 0; e < E0_; e++) s += bb0[((size_t)e * B_ + b) * OUT_ + j];
  for (int e = 0; e < E1_; e++) s += bb1[((size_t)e * B_ + b) * OUT_ + j];
  out[i] = 0.5f * s;
}

__global__ void scale_half_kernel(const float* __restrict__ X,
                                  float* __restrict__ out, long long n)
{
  long long i = (long long)blockIdx.x * blockDim.x + threadIdx.x;
  if (i < n) out[i] = 0.5f * X[i];
}

__global__ void lb_final_kernel(const float* __restrict__ lbacc,
                                float* __restrict__ out, int E, float inv)
{
  if (threadIdx.x == 0 && blockIdx.x == 0) {
    float s = 0.f;
    for (int e = 0; e < E; e++) s += (lbacc[e] * inv) * (lbacc[E + e] * inv);
    out[0] = (float)E * s;
  }
}

// ======================================================================
// host side
// ======================================================================
static inline void launch_gemm(hipStream_t st,
    const unsigned short* A, size_t sA, const unsigned short* W, size_t sW,
    const float* bias, size_t sBias, const float* R, size_t sR,
    float* C, size_t sC, int M, int N, int K, int E, int mode)
{
  dim3 grid((N + 255) / 256, (M + 31) / 32, E);
  wmma_gemm<<<grid, 256, 0, st>>>(A, sA, W, sW, bias, sBias, R, sR, C, sC, M, N, K, mode);
}

static inline void launch_gemm_tdm(hipStream_t st,
    const unsigned short* A, size_t sA, const unsigned short* W, size_t sW,
    const float* bias, size_t sBias, float* C, size_t sC,
    int M, int N, int K, int E, int mode)
{
  dim3 grid((N + 255) / 256, (M + 31) / 32, E);
  wmma_gemm_tdm<<<grid, 256, 0, st>>>(A, sA, W, sW, bias, sBias, C, sC, M, N, K, mode);
}

static inline void launch_transpose(hipStream_t st, const float* Wf, size_t sIn,
                                    unsigned short* Wt, size_t sOut,
                                    int K, int N, int Kp, int E)
{
  dim3 grid((N + 31) / 32, (Kp + 31) / 32, E);
  transpose_cvt<<<grid, 256, 0, st>>>(Wf, sIn, Wt, sOut, K, N, Kp);
}

extern "C" void kernel_launch(void* const* d_in, const int* in_sizes, int n_in,
                              void* d_out, int out_size, void* d_ws, size_t ws_size,
                              hipStream_t stream)
{
  (void)in_sizes; (void)n_in; (void)out_size; (void)ws_size;
  const float* voxel   = (const float*)d_in[0];
  const float* ridge_w = (const float*)d_in[1];
  const float* ridge_b = (const float*)d_in[2];
  const float* r0_gw   = (const float*)d_in[3];
  const float* r0_gb   = (const float*)d_in[4];
  const float* e0_ln_s = (const float*)d_in[5];
  const float* e0_ln_b = (const float*)d_in[6];
  const float* e0_W    = (const float*)d_in[7];
  const float* e0_b    = (const float*)d_in[8];
  const float* bp0_W   = (const float*)d_in[9];
  const float* bp0_b   = (const float*)d_in[10];
  const float* p0_g1 = (const float*)d_in[11]; const float* p0_b1 = (const float*)d_in[12];
  const float* p0_W1 = (const float*)d_in[13]; const float* p0_c1 = (const float*)d_in[14];
  const float* p0_g2 = (const float*)d_in[15]; const float* p0_b2 = (const float*)d_in[16];
  const float* p0_W2 = (const float*)d_in[17]; const float* p0_c2 = (const float*)d_in[18];
  const float* p0_g3 = (const float*)d_in[19]; const float* p0_b3 = (const float*)d_in[20];
  const float* p0_W3 = (const float*)d_in[21]; const float* p0_c3 = (const float*)d_in[22];
  const float* r1_gw   = (const float*)d_in[23];
  const float* r1_gb   = (const float*)d_in[24];
  const float* e1_W    = (const float*)d_in[25];
  const float* e1_b    = (const float*)d_in[26];
  const float* bp1_W   = (const float*)d_in[27];
  const float* bp1_b   = (const float*)d_in[28];
  const float* p1_g1 = (const float*)d_in[29]; const float* p1_b1 = (const float*)d_in[30];
  const float* p1_W1 = (const float*)d_in[31]; const float* p1_c1 = (const float*)d_in[32];
  const float* p1_g2 = (const float*)d_in[33]; const float* p1_b2 = (const float*)d_in[34];
  const float* p1_W2 = (const float*)d_in[35]; const float* p1_c2 = (const float*)d_in[36];
  const float* p1_g3 = (const float*)d_in[37]; const float* p1_b3 = (const float*)d_in[38];
  const float* p1_W3 = (const float*)d_in[39]; const float* p1_c3 = (const float*)d_in[40];
  float* out = (float*)d_out;

  char* base = (char*)d_ws; size_t off = 0;
  auto A256 = [&](size_t bytes) -> void* {
    void* p = base + off; off = (off + bytes + 255) & ~(size_t)255; return p;
  };
  unsigned short* voxel_bf = (unsigned short*)A256((size_t)B_ * KP_ * 2);
  unsigned short* ridgeWT  = (unsigned short*)A256((size_t)H_ * KP_ * 2);
  unsigned short* e0WT     = (unsigned short*)A256((size_t)E0_ * NB_ * H0_ * H0_ * 2);
  unsigned short* bp0WT    = (unsigned short*)A256((size_t)E0_ * OUT_ * H0_ * 2);
  unsigned short* c0W1T    = (unsigned short*)A256((size_t)E0_ * PH_ * D_ * 2);
  unsigned short* c0W2T    = (unsigned short*)A256((size_t)E0_ * PH_ * PH_ * 2);
  unsigned short* c0W3T    = (unsigned short*)A256((size_t)E0_ * D_ * PH_ * 2);
  unsigned short* e1WT     = (unsigned short*)A256((size_t)E1_ * H1_ * H1_ * 2);
  unsigned short* bp1WT    = (unsigned short*)A256((size_t)E1_ * OUT_ * H1_ * 2);
  unsigned short* c1W1T    = (unsigned short*)A256((size_t)E1_ * PH_ * D_ * 2);
  unsigned short* c1W2T    = (unsigned short*)A256((size_t)E1_ * PH_ * PH_ * 2);
  unsigned short* c1W3T    = (unsigned short*)A256((size_t)E1_ * D_ * PH_ * 2);
  float*          vr_f     = (float*)A256((size_t)B_ * H_ * 4);
  float*          sel0_f   = (float*)A256((size_t)E0_ * B_ * C0_ * 4);
  float*          w0_f     = (float*)A256((size_t)E0_ * B_ * 4);
  unsigned short* x_bf     = (unsigned short*)A256((size_t)E0_ * B_ * H0_ * 2);
  float*          xA       = (float*)A256((size_t)E0_ * B_ * H0_ * 4);
  float*          xB       = (float*)A256((size_t)E0_ * B_ * H0_ * 4);
  unsigned short* feat0_bf = (unsigned short*)A256((size_t)E0_ * B_ * H0_ * 2);
  float*          bb0_f    = (float*)A256((size_t)E0_ * B_ * OUT_ * 4);
  float*          sel1_f   = (float*)A256((size_t)E1_ * B_ * C1_ * 4);
  float*          w1_f     = (float*)A256((size_t)E1_ * B_ * 4);
  unsigned short* x1_bf    = (unsigned short*)A256((size_t)E1_ * B_ * H1_ * 2);
  float*          feat1_f  = (float*)A256((size_t)E1_ * B_ * H1_ * 4);
  unsigned short* feat1_bf = (unsigned short*)A256((size_t)E1_ * B_ * H1_ * 2);
  float*          bb1_f    = (float*)A256((size_t)E1_ * B_ * OUT_ * 4);
  unsigned short* pre_bf   = (unsigned short*)A256((size_t)E1_ * B_ * S_ * PH_ * 2);
  float*          hA       = (float*)A256((size_t)E1_ * B_ * S_ * PH_ * 4);
  float*          clip_acc = (float*)A256((size_t)B_ * OUT_ * 4);
  float*          lbacc    = (float*)A256(64);

  const int ROWS = B_ * S_;                 // 2464 projector rows per expert
  const long long NBK = (long long)B_ * OUT_;

  hipMemsetAsync(sel0_f, 0, (size_t)E0_ * B_ * C0_ * 4, stream);
  hipMemsetAsync(sel1_f, 0, (size_t)E1_ * B_ * C1_ * 4, stream);
  hipMemsetAsync(clip_acc, 0, (size_t)B_ * OUT_ * 4, stream);
  hipMemsetAsync(lbacc, 0, 64, stream);

  // -------- weight prep: f32 -> bf16 (transposed [N,K]) --------
  {
    long long n = (long long)B_ * KP_;
    cvt_pad_kernel<<<dim3((unsigned)((n + 255) / 256)), 256, 0, stream>>>(voxel, voxel_bf, V_, KP_, n);
  }
  launch_transpose(stream, ridge_w, 0, ridgeWT, 0, V_, H_, KP_, 1);
  launch_transpose(stream, e0_W,  (size_t)H0_*H0_, e0WT,  (size_t)H0_*H0_, H0_, H0_, H0_, E0_*NB_);
  launch_transpose(stream, bp0_W, (size_t)H0_*OUT_, bp0WT, (size_t)OUT_*H0_, H0_, OUT_, H0_, E0_);
  launch_transpose(stream, p0_W1, (size_t)D_*PH_,  c0W1T, (size_t)PH_*D_,  D_,  PH_, D_,  E0_);
  launch_transpose(stream, p0_W2, (size_t)PH_*PH_, c0W2T, (size_t)PH_*PH_, PH_, PH_, PH_, E0_);
  launch_transpose(stream, p0_W3, (size_t)PH_*D_,  c0W3T, (size_t)D_*PH_,  PH_, D_,  PH_, E0_);
  launch_transpose(stream, e1_W,  (size_t)H1_*H1_, e1WT,  (size_t)H1_*H1_, H1_, H1_, H1_, E1_);
  launch_transpose(stream, bp1_W, (size_t)H1_*OUT_, bp1WT, (size_t)OUT_*H1_, H1_, OUT_, H1_, E1_);
  launch_transpose(stream, p1_W1, (size_t)D_*PH_,  c1W1T, (size_t)PH_*D_,  D_,  PH_, D_,  E1_);
  launch_transpose(stream, p1_W2, (size_t)PH_*PH_, c1W2T, (size_t)PH_*PH_, PH_, PH_, PH_, E1_);
  launch_transpose(stream, p1_W3, (size_t)PH_*D_,  c1W3T, (size_t)D_*PH_,  PH_, D_,  PH_, E1_);

  // -------- ridge (TDM-staged WMMA) --------
  launch_gemm_tdm(stream, voxel_bf, 0, ridgeWT, 0, ridge_b, 0,
                  vr_f, 0, B_, H_, KP_, 1, 0);

  // -------- level-0 router --------
  {
    size_t shmem = (size_t)H_ * E0_ * 4 + (size_t)H_ * 4;
    router_kernel<<<dim3(B_, 1), 256, shmem, stream>>>(
        vr_f, r0_gw, r0_gb, sel0_f, w0_f, lbacc, B_, H_, E0_, C0_);
  }
  {
    long long n = (long long)E0_ * B_ * C0_;
    cvt_scale_kernel<<<dim3((unsigned)((n + 255) / 256)), 256, 0, stream>>>(
        sel0_f, w0_f, x_bf, xA, n, C0_);
  }

  // -------- forward_feat: 4 residual LN->Linear->GELU blocks --------
  float* cur = xA; float* alt = xB;
  for (int k = 0; k < NB_; k++) {
    ln_gelu_kernel<<<dim3(E0_ * B_), 256, 0, stream>>>(
        cur, e0_ln_s + (size_t)k * H0_, e0_ln_b + (size_t)k * H0_,
        (size_t)NB_ * H0_, x_bf, H0_, B_, 0);
    launch_gemm(stream,
        x_bf, (size_t)B_ * H0_,
        e0WT + (size_t)k * H0_ * H0_, (size_t)NB_ * H0_ * H0_,
        e0_b + (size_t)k * H0_, (size_t)NB_ * H0_,
        cur, (size_t)B_ * H0_,
        alt, (size_t)B_ * H0_,
        B_, H0_, H0_, E0_, 1);
    float* t = cur; cur = alt; alt = t;
  }
  {
    long long n = (long long)E0_ * B_ * H0_;
    cvt_scale_kernel<<<dim3((unsigned)((n + 255) / 256)), 256, 0, stream>>>(
        cur, nullptr, feat0_bf, nullptr, n, H0_);
  }

  // -------- bb0 = feat0 @ bp0_W + bp0_b --------
  launch_gemm_tdm(stream, feat0_bf, (size_t)B_ * H0_, bp0WT, (size_t)OUT_ * H0_,
                  bp0_b, (size_t)OUT_, bb0_f, (size_t)B_ * OUT_,
                  B_, OUT_, H0_, E0_, 0);

  // -------- level-1 routers --------
  {
    size_t shmem = (size_t)H0_ * EF_ * 4 + (size_t)H0_ * 4;
    router_kernel<<<dim3(B_, E0_), 256, shmem, stream>>>(
        cur, r1_gw, r1_gb, sel1_f, w1_f, nullptr, B_, H0_, EF_, C1_);
  }
  {
    long long n = (long long)E1_ * B_ * C1_;
    cvt_scale_kernel<<<dim3((unsigned)((n + 255) / 256)), 256, 0, stream>>>(
        sel1_f, w1_f, x1_bf, nullptr, n, C1_);
  }
  launch_gemm_tdm(stream, x1_bf, (size_t)B_ * H1_, e1WT, (size_t)H1_ * H1_,
                  e1_b, (size_t)H1_, feat1_f, (size_t)B_ * H1_,
                  B_, H1_, H1_, E1_, 0);
  {
    long long n = (long long)E1_ * B_ * H1_;
    cvt_scale_kernel<<<dim3((unsigned)((n + 255) / 256)), 256, 0, stream>>>(
        feat1_f, nullptr, feat1_bf, nullptr, n, H1_);
  }
  launch_gemm_tdm(stream, feat1_bf, (size_t)B_ * H1_, bp1WT, (size_t)OUT_ * H1_,
                  bp1_b, (size_t)OUT_, bb1_f, (size_t)B_ * OUT_,
                  B_, OUT_, H1_, E1_, 0);

  // -------- projector level-0 --------
  ln_gelu_kernel<<<dim3(E0_ * ROWS), 256, 0, stream>>>(
      bb0_f, p0_g1, p0_b1, (size_t)D_, pre_bf, D_, ROWS, 1);
  launch_gemm_tdm(stream, pre_bf, (size_t)ROWS * D_, c0W1T, (size_t)PH_ * D_,
                  p0_c1, (size_t)PH_, hA, (size_t)ROWS * PH_,
                  ROWS, PH_, D_, E0_, 0);
  ln_gelu_kernel<<<dim3(E0_ * ROWS), 256, 0, stream>>>(
      hA, p0_g2, p0_b2, (size_t)PH_, pre_bf, PH_, ROWS, 1);
  launch_gemm_tdm(stream, pre_bf, (size_t)ROWS * PH_, c0W2T, (size_t)PH_ * PH_,
                  p0_c2, (size_t)PH_, hA, (size_t)ROWS * PH_,
                  ROWS, PH_, PH_, E0_, 0);
  ln_gelu_kernel<<<dim3(E0_ * ROWS), 256, 0, stream>>>(
      hA, p0_g3, p0_b3, (size_t)PH_, pre_bf, PH_, ROWS, 1);
  launch_gemm_tdm(stream, pre_bf, (size_t)ROWS * PH_, c0W3T, (size_t)D_ * PH_,
                  p0_c3, (size_t)D_, clip_acc, 0,
                  ROWS, D_, PH_, E0_, 2);

  // -------- projector level-1 --------
  ln_gelu_kernel<<<dim3(E1_ * ROWS), 256, 0, stream>>>(
      bb1_f, p1_g1, p1_b1, (size_t)D_, pre_bf, D_, ROWS, 1);
  launch_gemm_tdm(stream, pre_bf, (size_t)ROWS * D_, c1W1T, (size_t)PH_ * D_,
                  p1_c1, (size_t)PH_, hA, (size_t)ROWS * PH_,
                  ROWS, PH_, D_, E1_, 0);
  ln_gelu_kernel<<<dim3(E1_ * ROWS), 256, 0, stream>>>(
      hA, p1_g2, p1_b2, (size_t)PH_, pre_bf, PH_, ROWS, 1);
  launch_gemm_tdm(stream, pre_bf, (size_t)ROWS * PH_, c1W2T, (size_t)PH_ * PH_,
                  p1_c2, (size_t)PH_, hA, (size_t)ROWS * PH_,
                  ROWS, PH_, PH_, E1_, 0);
  ln_gelu_kernel<<<dim3(E1_ * ROWS), 256, 0, stream>>>(
      hA, p1_g3, p1_b3, (size_t)PH_, pre_bf, PH_, ROWS, 1);
  launch_gemm_tdm(stream, pre_bf, (size_t)ROWS * PH_, c1W3T, (size_t)D_ * PH_,
                  p1_c3, (size_t)D_, clip_acc, 0,
                  ROWS, D_, PH_, E1_, 2);

  // -------- outputs --------
  combine_backbone_kernel<<<dim3((unsigned)((NBK + 255) / 256)), 256, 0, stream>>>(
      bb0_f, bb1_f, out, NBK);
  scale_half_kernel<<<dim3((unsigned)((NBK + 255) / 256)), 256, 0, stream>>>(
      clip_acc, out + NBK, NBK);
  lb_final_kernel<<<dim3(1), 32, 0, stream>>>(
      lbacc, out + 2 * NBK, E0_, 1.0f / ((float)B_ * (float)H_));
}
